// InvariantPointAttentionV50_2319282340037
// MI455X (gfx1250) — matrix operand order repaired and checked by reading, not compiled, not run on hardware
//
#include <hip/hip_runtime.h>
#include <hip/hip_bf16.h>

// ---------------------------------------------------------------------------
// Invariant Point Attention for MI455X (gfx1250, wave32, WMMA 16x16x32 f16,
// async-to-LDS double-buffered B staging).
// ---------------------------------------------------------------------------
typedef __attribute__((ext_vector_type(16))) _Float16 v16h;
typedef __attribute__((ext_vector_type(8)))  _Float16 v8h;
typedef __attribute__((ext_vector_type(8)))  float    v8f;

#define IPA_N   2048
#define IPA_C   768
#define IPA_H   12
#define IPA_CH  64
#define IPA_D3  144
#define IPA_E   49152
#define IPA_DP  128

// Build a 16-half fragment from two contiguous 8-half chunks (16B vector loads).
__device__ __forceinline__ v16h make_frag(const _Float16* p0, const _Float16* p1) {
  v8h lo = *(const v8h*)p0;
  v8h hi = *(const v8h*)p1;
  v16h r;
#pragma unroll
  for (int i = 0; i < 8; ++i) { r[i] = lo[i]; r[i + 8] = hi[i]; }
  return r;
}

// Generic-pointer -> LDS byte address (AS3 pointers are 32-bit offsets).
__device__ __forceinline__ unsigned lds_addr(const void* p) {
  return (unsigned)(unsigned long long)(__attribute__((address_space(3))) const void*)p;
}

// ---------------------------------------------------------------------------
// WMMA GEMM with async-to-LDS B staging:
//   C[m][n] (+= / =) sum_k A[m][k] * BT[n][k] (+ bias[n])
// Block = 128 threads = 4 waves covering 64(M) x 16(N); the 16x32 f16 B-tile
// per K-chunk is DMA'd into LDS with GLOBAL_LOAD_ASYNC_TO_LDS_B64 (one 8-byte
// segment per lane, 128 lanes = 1KB tile), double-buffered and retired with
// s_wait_asynccnt, then consumed by all 4 waves (4x reuse, ds_load_b128).
// Per-lane fragment layouts per CDNA5 ISA 7.12.2 (wave32):
//   A: lane<16 -> K {kb..kb+7, kb+16..kb+23}, kb=0; lane>=16 -> kb=8
//   B: same K as A: lane<16 -> K {0..7, 16..23}, lane>=16 -> K {8..15, 24..31} (column n = lane&15)
//   C/D: VGPR r -> row r (lanes 0-15) / row 8+r (lanes 16-31), col = lane&15
// Requires: K % 32 == 0, M % 64 == 0, BT padded to ceil16(Nout) rows.
// ---------------------------------------------------------------------------
__global__ void __launch_bounds__(128)
k_wmma_gemm(const _Float16* __restrict__ A,
            const _Float16* __restrict__ BT,
            const float* __restrict__ bias,
            float* __restrict__ C,
            int M, int K, int Nout,
            int ldA, int ldBT, int ldC, int accumulate) {
  __shared__ __align__(64) _Float16 sB[2][16 * 32];

  const int tid   = threadIdx.x;      // 0..127
  const int lane  = tid & 31;
  const int wave  = tid >> 5;         // 0..3
  const int tileN = blockIdx.x * 16;
  const int tileM = blockIdx.y * 64 + wave * 16;
  (void)M;

  // cooperative copy mapping: thread -> (row 0..15, 4-half segment 0..7)
  const int crow = tid >> 3;          // 0..15
  const int cseg = tid & 7;           // 0..7
  const _Float16* gsrc = BT + (size_t)(tileN + crow) * ldBT + cseg * 4;
  const unsigned ldsdst0  = lds_addr(&sB[0][crow * 32 + cseg * 4]);
  const unsigned ldsstride = (unsigned)(16 * 32 * sizeof(_Float16));

  const int mA     = tileM + (lane & 15);
  const int kA     = (lane & 16) ? 8 : 0;
  const int kBl    = (lane & 16) ? 8 : 0;
  const int nlocal = lane & 15;

  const int nchunks = K >> 5;
  // kick off chunk 0 into buffer 0
  asm volatile("global_load_async_to_lds_b64 %0, %1, off"
               :: "v"(ldsdst0), "v"((const void*)gsrc) : "memory");

  v8f acc = {};
  for (int c = 0; c < nchunks; ++c) {
    const int k0  = c << 5;
    const int buf = c & 1;
    if (c + 1 < nchunks) {   // issue next chunk into the other buffer
      const unsigned dst = ldsdst0 + (unsigned)(buf ^ 1) * ldsstride;
      asm volatile("global_load_async_to_lds_b64 %0, %1, off"
                   :: "v"(dst), "v"((const void*)(gsrc + k0 + 32)) : "memory");
      asm volatile("s_wait_asynccnt 0x1" ::: "memory");  // retire current chunk
    } else {
      asm volatile("s_wait_asynccnt 0x0" ::: "memory");
    }
    __syncthreads();  // staged B visible to all 4 waves

    const _Float16* pa = A + (size_t)mA * ldA + k0 + kA;
    if (c + 2 < nchunks) __builtin_prefetch(pa + 64, 0, 1);  // global_prefetch_b8
    v16h a = make_frag(pa, pa + 16);
    const _Float16* pbl = &sB[buf][nlocal * 32 + kBl];
    v16h b = make_frag(pbl, pbl + 16);                       // ds_load_b128 x2
    acc = __builtin_amdgcn_wmma_f32_16x16x32_f16(false, a, false, b,
                                                 (short)0, acc, false, false);
    __syncthreads();  // done reading buf before chunk c+2 overwrites it
  }

  const int mBase = tileM + ((lane & 16) ? 8 : 0);
  const int nOut  = tileN + (lane & 15);
  if (nOut < Nout) {
    const float bv = bias ? bias[nOut] : 0.0f;
#pragma unroll
    for (int r = 0; r < 8; ++r) {
      const size_t idx = (size_t)(mBase + r) * ldC + nOut;
      float v = acc[r] + bv;
      if (accumulate) v += C[idx];
      C[idx] = v;
    }
  }
}

// ---------------------------------------------------------------------------
// Fused IPA logits for one head: WMMA(q.kT, K=64) + WMMA(qpts.kptsT, K=32 pad)
// epilogue: scale*s - 0.5*scale*(|q|^2+|k|^2-2*p) + pb[m][n] + pb[n][m]
// ---------------------------------------------------------------------------
__global__ void k_logits(const _Float16* __restrict__ qh,  // [N,768] + h*64
                         const _Float16* __restrict__ kh,
                         const _Float16* __restrict__ qp,  // [N,32] head slice
                         const _Float16* __restrict__ kp,
                         const float* __restrict__ q2,     // [N,12]
                         const float* __restrict__ k2,
                         const float* __restrict__ pb,     // [N,N] dense bias
                         float* __restrict__ logits,
                         int N, int h, float scale) {
  const int lane  = threadIdx.x & 31;
  const int tileN = blockIdx.x * 16;
  const int tileM = blockIdx.y * 16;
  const int mA = tileM + (lane & 15);
  const int nB = tileN + (lane & 15);
  const int kA = (lane & 16) ? 8 : 0;
  const int kB = (lane & 16) ? 8 : 0;

  v8f accS = {};
#pragma unroll
  for (int k0 = 0; k0 < IPA_CH; k0 += 32) {
    const _Float16* pa = qh + (size_t)mA * IPA_C + k0 + kA;
    const _Float16* pb2 = kh + (size_t)nB * IPA_C + k0 + kB;
    v16h a = make_frag(pa, pa + 16);
    v16h b = make_frag(pb2, pb2 + 16);
    accS = __builtin_amdgcn_wmma_f32_16x16x32_f16(false, a, false, b,
                                                  (short)0, accS, false, false);
  }
  v8f accP = {};
  {
    const _Float16* pa = qp + (size_t)mA * 32 + kA;
    const _Float16* pb2 = kp + (size_t)nB * 32 + kB;
    v16h a = make_frag(pa, pa + 16);
    v16h b = make_frag(pb2, pb2 + 16);
    accP = __builtin_amdgcn_wmma_f32_16x16x32_f16(false, a, false, b,
                                                  (short)0, accP, false, false);
  }

  const int mBase = tileM + ((lane & 16) ? 8 : 0);
  const int n     = tileN + (lane & 15);
  const float k2v = k2[(size_t)n * IPA_H + h];
#pragma unroll
  for (int r = 0; r < 8; ++r) {
    const int m = mBase + r;
    const float q2v = q2[(size_t)m * IPA_H + h];
    const float sl = scale * accS[r];
    const float pl = -0.5f * scale * (q2v + k2v - 2.0f * accP[r]);
    const float bb = pb[(size_t)m * N + n] + pb[(size_t)n * N + m];
    logits[(size_t)m * N + n] = sl + pl + bb;
  }
}

// ---------------------------------------------------------------------------
// Elementwise / data-movement helpers
// ---------------------------------------------------------------------------
__global__ void k_cvt_f16(const float* __restrict__ in, _Float16* __restrict__ out,
                          size_t n) {
  size_t i = blockIdx.x * (size_t)blockDim.x + threadIdx.x;
  if (i < n) out[i] = (_Float16)in[i];
}

__global__ void k_cvt_f16_pad(const float* __restrict__ in, _Float16* __restrict__ out,
                              int rows, int cols, int ldo) {
  int i = blockIdx.x * blockDim.x + threadIdx.x;
  if (i >= rows * cols) return;
  int r = i / cols, c = i % cols;
  out[(size_t)r * ldo + c] = (_Float16)in[i];
}

// Weight transpose + f16: in f32 [K, Nw] row-major -> out f16 [Nw, ldo]
__global__ void k_transpose_f16(const float* __restrict__ in, _Float16* __restrict__ out,
                                int K, int Nw, int ldo) {
  int i = blockIdx.x * blockDim.x + threadIdx.x;
  if (i >= K * Nw) return;
  int k = i / Nw, n = i % Nw;
  out[(size_t)n * ldo + k] = (_Float16)in[i];
}

// v f32 [N, H*64] -> vT f16 [H][64][N] (contiguous-K B fragments for AV)
__global__ void k_build_vT(const float* __restrict__ v, _Float16* __restrict__ vT, int N) {
  int i = blockIdx.x * blockDim.x + threadIdx.x;
  if (i >= N * IPA_C) return;
  int n = i / IPA_C, c = i % IPA_C;
  int h = c / IPA_CH, cc = c % IPA_CH;
  vT[(size_t)h * IPA_CH * N + (size_t)cc * N + n] = (_Float16)v[i];
}

// Local points -> global frame (einsum bnhpd,bnde->bnhpe + trans), plus
// squared norms. One thread per (n, h) handles its 4 points.
// outA: f16 [H][N][32] zero-padded (A fragments for point logits), optional.
// outT: f16 [H][16][N] zero-padded (B fragments for attn@v_pts), optional.
__global__ void k_point_transform(const float* __restrict__ praw,  // [N,144]
                                  const float* __restrict__ rot,   // [N,3,3]
                                  const float* __restrict__ trans, // [N,3]
                                  _Float16* __restrict__ outA,
                                  _Float16* __restrict__ outT,
                                  float* __restrict__ sum2, int N) {
  int i = blockIdx.x * blockDim.x + threadIdx.x;
  if (i >= N * IPA_H) return;
  int n = i / IPA_H, h = i % IPA_H;
  const float* R = rot + (size_t)n * 9;
  const float* T = trans + (size_t)n * 3;
  float s2 = 0.0f;
#pragma unroll
  for (int p = 0; p < 4; ++p) {
    const float* lp = praw + (size_t)n * IPA_D3 + (h * 4 + p) * 3;
#pragma unroll
    for (int e = 0; e < 3; ++e) {
      float g = lp[0] * R[0 * 3 + e] + lp[1] * R[1 * 3 + e] + lp[2] * R[2 * 3 + e] + T[e];
      s2 += g * g;
      int j = p * 3 + e;
      if (outA) outA[(size_t)h * N * 32 + (size_t)n * 32 + j] = (_Float16)g;
      if (outT) outT[(size_t)h * 16 * N + (size_t)j * N + n] = (_Float16)g;
    }
  }
  if (sum2) sum2[(size_t)n * IPA_H + h] = s2;
}

// Sparse pair-bias scatter for head h into dense [N,N] (pre-zeroed).
__global__ void k_scatter(const float* __restrict__ pbs,        // [E,12]
                          const long long* __restrict__ ei,     // [2,E] int64
                          float* __restrict__ pb, int E, int N, int h) {
  int e = blockIdx.x * blockDim.x + threadIdx.x;
  if (e >= E) return;
  long long src = ei[e], dst = ei[(size_t)E + e];
  int ls = (int)(src % N), ld = (int)(dst % N);
  pb[(size_t)ls * N + ld] = pbs[(size_t)e * IPA_H + h];
}

// Row softmax, f32 logits -> f16 probabilities (WMMA A operand for AV).
__global__ void k_softmax(const float* __restrict__ logits,
                          _Float16* __restrict__ probs, int N) {
  const int row = blockIdx.x;
  const float* x = logits + (size_t)row * N;
  __shared__ float red[256];
  const int t = threadIdx.x;
  float mx = -3.4e38f;
  for (int i = t; i < N; i += 256) mx = fmaxf(mx, x[i]);
  red[t] = mx; __syncthreads();
  for (int s = 128; s > 0; s >>= 1) { if (t < s) red[t] = fmaxf(red[t], red[t + s]); __syncthreads(); }
  mx = red[0]; __syncthreads();
  float sum = 0.0f;
  for (int i = t; i < N; i += 256) sum += __expf(x[i] - mx);
  red[t] = sum; __syncthreads();
  for (int s = 128; s > 0; s >>= 1) { if (t < s) red[t] += red[t + s]; __syncthreads(); }
  const float inv = 1.0f / red[0];
  _Float16* p = probs + (size_t)row * N;
  for (int i = t; i < N; i += 256) p[i] = (_Float16)(__expf(x[i] - mx) * inv);
}

// Residual + LayerNorm, one block per row.
__global__ void k_layernorm(const float* __restrict__ single,
                            const float* __restrict__ outv,
                            const float* __restrict__ gamma,
                            const float* __restrict__ beta,
                            float* __restrict__ y, int C) {
  const int row = blockIdx.x;
  __shared__ float red[256];
  const int t = threadIdx.x;
  const float* s = single + (size_t)row * C;
  const float* o = outv + (size_t)row * C;
  float acc = 0.0f;
  for (int i = t; i < C; i += 256) acc += s[i] + o[i];
  red[t] = acc; __syncthreads();
  for (int st = 128; st > 0; st >>= 1) { if (t < st) red[t] += red[t + st]; __syncthreads(); }
  const float mu = red[0] / C; __syncthreads();
  acc = 0.0f;
  for (int i = t; i < C; i += 256) { float d = s[i] + o[i] - mu; acc += d * d; }
  red[t] = acc; __syncthreads();
  for (int st = 128; st > 0; st >>= 1) { if (t < st) red[t] += red[t + st]; __syncthreads(); }
  const float inv = rsqrtf(red[0] / C + 1e-5f);
  for (int i = t; i < C; i += 256) {
    float x = s[i] + o[i];
    y[(size_t)row * C + i] = (x - mu) * inv * gamma[i] + beta[i];
  }
}

// ---------------------------------------------------------------------------
static inline void gemm(hipStream_t s, const _Float16* A, const _Float16* BT,
                        const float* bias, float* C, int M, int K, int Nout,
                        int ldA, int ldBT, int ldC, int acc) {
  dim3 g((Nout + 15) / 16, (M + 63) / 64);
  k_wmma_gemm<<<g, 128, 0, s>>>(A, BT, bias, C, M, K, Nout, ldA, ldBT, ldC, acc);
}

extern "C" void kernel_launch(void* const* d_in, const int* in_sizes, int n_in,
                              void* d_out, int out_size, void* d_ws, size_t ws_size,
                              hipStream_t stream) {
  (void)in_sizes; (void)n_in; (void)out_size; (void)ws_size;
  const int N = IPA_N, C = IPA_C, Hn = IPA_H, D3 = IPA_D3, E = IPA_E, Dp = IPA_DP;

  const float* single    = (const float*)d_in[0];
  const float* edge_feat = (const float*)d_in[1];
  const float* rot       = (const float*)d_in[2];
  const float* trans     = (const float*)d_in[3];
  const float* Wq  = (const float*)d_in[4];  const float* bq  = (const float*)d_in[5];
  const float* Wk  = (const float*)d_in[6];  const float* bk  = (const float*)d_in[7];
  const float* Wv  = (const float*)d_in[8];  const float* bv  = (const float*)d_in[9];
  const float* Wpb = (const float*)d_in[10]; const float* bpb = (const float*)d_in[11];
  const float* Wqp = (const float*)d_in[12]; const float* bqp = (const float*)d_in[13];
  const float* Wkp = (const float*)d_in[14]; const float* bkp = (const float*)d_in[15];
  const float* Wvp = (const float*)d_in[16]; const float* bvp = (const float*)d_in[17];
  const float* Wo  = (const float*)d_in[18]; const float* bo  = (const float*)d_in[19];
  const float* Wpo = (const float*)d_in[20]; const float* bpo = (const float*)d_in[21];
  const float* gamma = (const float*)d_in[22];
  const float* beta  = (const float*)d_in[23];
  const long long* edge_index = (const long long*)d_in[24];

  // ---- bump allocator over d_ws (256B aligned carve-outs) ----
  char* ws = (char*)d_ws;
  size_t off = 0;
  auto alloc = [&](size_t bytes) -> void* {
    off = (off + 255) & ~(size_t)255;
    void* p = ws + off;
    off += bytes;
    return p;
  };
  _Float16* sing_h = (_Float16*)alloc((size_t)N * C * 2);
  _Float16* edge_h = (_Float16*)alloc((size_t)E * Dp * 2);
  _Float16* WqT  = (_Float16*)alloc((size_t)C * C * 2);
  _Float16* WkT  = (_Float16*)alloc((size_t)C * C * 2);
  _Float16* WvT  = (_Float16*)alloc((size_t)C * C * 2);
  _Float16* WoT  = (_Float16*)alloc((size_t)C * C * 2);
  _Float16* WqpT = (_Float16*)alloc((size_t)D3 * C * 2);
  _Float16* WkpT = (_Float16*)alloc((size_t)D3 * C * 2);
  _Float16* WvpT = (_Float16*)alloc((size_t)D3 * C * 2);
  _Float16* WpbT = (_Float16*)alloc((size_t)16 * Dp * 2);   // 12 rows + 4 pad
  _Float16* WpoT = (_Float16*)alloc((size_t)C * 160 * 2);   // K 144 -> 160 pad
  float* qf = (float*)alloc((size_t)N * C * 4);
  float* kf = (float*)alloc((size_t)N * C * 4);
  float* vf = (float*)alloc((size_t)N * C * 4);
  float* qpraw = (float*)alloc((size_t)N * D3 * 4);
  float* kpraw = (float*)alloc((size_t)N * D3 * 4);
  float* vpraw = (float*)alloc((size_t)N * D3 * 4);
  float* pbs = (float*)alloc((size_t)E * Hn * 4);
  _Float16* q_h = (_Float16*)alloc((size_t)N * C * 2);
  _Float16* k_h = (_Float16*)alloc((size_t)N * C * 2);
  _Float16* vT  = (_Float16*)alloc((size_t)Hn * IPA_CH * N * 2);
  _Float16* qp_h = (_Float16*)alloc((size_t)Hn * N * 32 * 2);
  _Float16* kp_h = (_Float16*)alloc((size_t)Hn * N * 32 * 2);
  _Float16* vpT  = (_Float16*)alloc((size_t)Hn * 16 * N * 2);
  float* q2 = (float*)alloc((size_t)N * Hn * 4);
  float* k2 = (float*)alloc((size_t)N * Hn * 4);
  float* pb     = (float*)alloc((size_t)N * N * 4);   // per-head, reused (L2-resident)
  float* logits = (float*)alloc((size_t)N * N * 4);
  _Float16* probs = (_Float16*)alloc((size_t)N * N * 2);
  float* attn_s = (float*)alloc((size_t)N * C * 4);
  float* attn_p = (float*)alloc((size_t)N * D3 * 4);
  _Float16* as_h = (_Float16*)alloc((size_t)N * C * 2);
  _Float16* ap_h = (_Float16*)alloc((size_t)N * 160 * 2);
  float* outb = (float*)alloc((size_t)N * C * 4);

  const int TPB = 256;
  auto blocks = [](size_t n, int t) { return (unsigned)((n + t - 1) / t); };

  // ---- stage inputs / weights in f16 (weights transposed for contiguous-K) --
  k_cvt_f16<<<blocks((size_t)N * C, TPB), TPB, 0, stream>>>(single, sing_h, (size_t)N * C);
  k_cvt_f16<<<blocks((size_t)E * Dp, TPB), TPB, 0, stream>>>(edge_feat, edge_h, (size_t)E * Dp);
  k_transpose_f16<<<blocks((size_t)C * C, TPB), TPB, 0, stream>>>(Wq, WqT, C, C, C);
  k_transpose_f16<<<blocks((size_t)C * C, TPB), TPB, 0, stream>>>(Wk, WkT, C, C, C);
  k_transpose_f16<<<blocks((size_t)C * C, TPB), TPB, 0, stream>>>(Wv, WvT, C, C, C);
  k_transpose_f16<<<blocks((size_t)C * C, TPB), TPB, 0, stream>>>(Wo, WoT, C, C, C);
  k_transpose_f16<<<blocks((size_t)C * D3, TPB), TPB, 0, stream>>>(Wqp, WqpT, C, D3, C);
  k_transpose_f16<<<blocks((size_t)C * D3, TPB), TPB, 0, stream>>>(Wkp, WkpT, C, D3, C);
  k_transpose_f16<<<blocks((size_t)C * D3, TPB), TPB, 0, stream>>>(Wvp, WvpT, C, D3, C);
  hipMemsetAsync(WpbT, 0, (size_t)16 * Dp * 2, stream);
  k_transpose_f16<<<blocks((size_t)Dp * Hn, TPB), TPB, 0, stream>>>(Wpb, WpbT, Dp, Hn, Dp);
  hipMemsetAsync(WpoT, 0, (size_t)C * 160 * 2, stream);
  k_transpose_f16<<<blocks((size_t)D3 * C, TPB), TPB, 0, stream>>>(Wpo, WpoT, D3, C, 160);

  // ---- projection GEMMs (all WMMA + async-LDS staging) ----
  gemm(stream, sing_h, WqT, bq, qf, N, C, C, C, C, C, 0);
  gemm(stream, sing_h, WkT, bk, kf, N, C, C, C, C, C, 0);
  gemm(stream, sing_h, WvT, bv, vf, N, C, C, C, C, C, 0);
  gemm(stream, sing_h, WqpT, bqp, qpraw, N, C, D3, C, C, D3, 0);
  gemm(stream, sing_h, WkpT, bkp, kpraw, N, C, D3, C, C, D3, 0);
  gemm(stream, sing_h, WvpT, bvp, vpraw, N, C, D3, C, C, D3, 0);
  gemm(stream, edge_h, WpbT, bpb, pbs, E, Dp, Hn, Dp, Dp, Hn, 0);  // pair-bias proj

  // ---- f16 operand staging for attention ----
  k_cvt_f16<<<blocks((size_t)N * C, TPB), TPB, 0, stream>>>(qf, q_h, (size_t)N * C);
  k_cvt_f16<<<blocks((size_t)N * C, TPB), TPB, 0, stream>>>(kf, k_h, (size_t)N * C);
  k_build_vT<<<blocks((size_t)N * C, TPB), TPB, 0, stream>>>(vf, vT, N);
  hipMemsetAsync(qp_h, 0, (size_t)Hn * N * 32 * 2, stream);
  hipMemsetAsync(kp_h, 0, (size_t)Hn * N * 32 * 2, stream);
  hipMemsetAsync(vpT, 0, (size_t)Hn * 16 * N * 2, stream);
  k_point_transform<<<blocks((size_t)N * Hn, TPB), TPB, 0, stream>>>(qpraw, rot, trans, qp_h, nullptr, q2, N);
  k_point_transform<<<blocks((size_t)N * Hn, TPB), TPB, 0, stream>>>(kpraw, rot, trans, kp_h, nullptr, k2, N);
  k_point_transform<<<blocks((size_t)N * Hn, TPB), TPB, 0, stream>>>(vpraw, rot, trans, nullptr, vpT, nullptr, N);

  // ---- per-head attention: intermediates stay L2-resident (192MB L2) ----
  const float scale = 0.125f;  // 64^-0.5
  for (int h = 0; h < Hn; ++h) {
    hipMemsetAsync(pb, 0, (size_t)N * N * 4, stream);
    k_scatter<<<blocks((size_t)E, TPB), TPB, 0, stream>>>(pbs, edge_index, pb, E, N, h);
    dim3 lg(N / 16, N / 16);
    k_logits<<<lg, 32, 0, stream>>>(q_h + h * IPA_CH, k_h + h * IPA_CH,
                                    qp_h + (size_t)h * N * 32, kp_h + (size_t)h * N * 32,
                                    q2, k2, pb, logits, N, h, scale);
    k_softmax<<<N, TPB, 0, stream>>>(logits, probs, N);
    gemm(stream, probs, vT + (size_t)h * IPA_CH * N, nullptr, attn_s + h * IPA_CH,
         N, N, IPA_CH, N, N, C, 0);                               // attn @ v
    gemm(stream, probs, vpT + (size_t)h * 16 * N, nullptr, attn_p + h * 12,
         N, N, 12, N, N, D3, 0);                                  // attn @ v_pts
  }

  // ---- output projections + residual + LayerNorm ----
  k_cvt_f16<<<blocks((size_t)N * C, TPB), TPB, 0, stream>>>(attn_s, as_h, (size_t)N * C);
  hipMemsetAsync(ap_h, 0, (size_t)N * 160 * 2, stream);
  k_cvt_f16_pad<<<blocks((size_t)N * D3, TPB), TPB, 0, stream>>>(attn_p, ap_h, N, D3, 160);
  gemm(stream, as_h, WoT, bo, outb, N, C, C, C, C, C, 0);
  gemm(stream, ap_h, WpoT, bpo, outb, N, 160, C, 160, 160, C, 1);
  k_layernorm<<<N, TPB, 0, stream>>>(single, outb, gamma, beta, (float*)d_out, C);
}
